// HAINT_Attention_77094662963332
// MI455X (gfx1250) — matrix operand order repaired and checked by reading, not compiled, not run on hardware
//
#include <hip/hip_runtime.h>
#include <hip/hip_fp16.h>

typedef __attribute__((ext_vector_type(16))) _Float16 v16h;
typedef __attribute__((ext_vector_type(8)))  _Float16 v8h;
typedef __attribute__((ext_vector_type(8)))  float    v8f;

#define Hdim 512
#define Tdim 512
#define Bdim 256
#define CROWS  128          // t-rows per workgroup chunk
#define RFRAG  8            // CROWS / 16 row fragments
#define NCHUNK 4            // Tdim / CROWS
#define NTW    2            // column tiles per wave (16 waves x 2 = 32)

// branch-free tanh: exact enough for f16 output, pure VALU (mul/exp/rcp)
__device__ __forceinline__ float fast_tanh(float x) {
    x = fminf(fmaxf(x, -9.0f), 9.0f);
    const float e = __expf(2.0f * x);                 // v_mul + v_exp_f32
    return (e - 1.0f) * __builtin_amdgcn_rcpf(e + 1.0f);
}

// 8-half-block swizzle for the x tile: keeps 16B A-chunks contiguous while
// spreading same-column accesses of 16 consecutive rows across all 64 banks.
__device__ __forceinline__ int xswz(int row, int off) {
    return row * Hdim + (off & 7) + ((((off >> 3) + row) & 63) << 3);
}

// ---------------------------------------------------------------------------
// Phase 1: s[b,k] = concat(ht,ct)[b,:] @ W_as[:,k] + ba[k]
// ---------------------------------------------------------------------------
__global__ void attn_phase1(const float* __restrict__ ht, const float* __restrict__ ct,
                            const float* __restrict__ W_as, const float* __restrict__ ba,
                            float* __restrict__ s_out) {
    const int b = blockIdx.x;
    const int k = threadIdx.x;              // 0..511
    const float* htb = ht + b * Hdim;
    const float* ctb = ct + b * Hdim;
    float acc = ba[k];
    for (int h = 0; h < Hdim; ++h) acc = fmaf(htb[h], W_as[h * Hdim + k], acc);
    const float* Wc = W_as + Hdim * Hdim;
    for (int h = 0; h < Hdim; ++h) acc = fmaf(ctb[h], Wc[h * Hdim + k], acc);
    s_out[b * Hdim + k] = acc;
}

// ---------------------------------------------------------------------------
// Pack W[h][k] (f32 row-major) into the per-lane f16 WMMA B-operand layout:
//   element j of lane l: K = 32*ks + (l<16 ? 0 : 8) + (j<8 ? j : 8+j), N = 16*nt + (l&15)
// ---------------------------------------------------------------------------
__global__ void attn_pack_w(const float* __restrict__ W, _Float16* __restrict__ Wpk) {
    const int idx  = blockIdx.x * blockDim.x + threadIdx.x;
    const int j    = idx & 15;
    const int lane = (idx >> 4) & 31;
    const int nt   = (idx >> 9) & 31;
    const int ks   = idx >> 14;
    const int h = ks * 32 + ((lane & 16) ? 8 : 0) + ((j < 8) ? j : (8 + j));
    const int k = nt * 16 + (lane & 15);
    Wpk[idx] = (_Float16)W[h * Hdim + k];
}

// ---------------------------------------------------------------------------
// Main fused chunk kernel: WG = (b, 128-row t-chunk), 512 threads (16 waves).
// Wave owns 2 column tiles x 8 row fragments (acc = 128 VGPRs, < 256 so no
// s_set_vgpr_msb in the hot loop). GEMM1 (A from global hi, cvt->f16) ->
// tanh -> x in LDS (128 KB) -> GEMM2 -> partial softmax state per (b,k,chunk).
// Each B-fragment fetch is amortized over 128 rows (8 wmma per fragment).
// ---------------------------------------------------------------------------
__launch_bounds__(512)
__global__ void attn_chunk(const float* __restrict__ hi,
                           const _Float16* __restrict__ W1pk,
                           const _Float16* __restrict__ W2pk,
                           const float* __restrict__ s_in,
                           float* __restrict__ cs_m,
                           float* __restrict__ cs_d,
                           float* __restrict__ cs_n) {
    __shared__ _Float16 x_lds[CROWS * Hdim];          // 128 KB (gfx1250: 320 KB/WG)

    const int wg   = blockIdx.x;              // 0..1023
    const int b    = wg >> 2;
    const int R0   = (wg & 3) * CROWS;        // first t-row of this chunk
    const int tid  = threadIdx.x;
    const int lane = tid & 31;
    const int wave = tid >> 5;                // 0..15 ; owns nt = wave + 16*i

    // A-operand geometry (16-bit A layout, wave32)
    const int arow  = lane & 15;        // row within 16-row fragment
    const int asel  = (lane >> 4) * 8;  // K sub-offset: 0 or 8
    const int rbase = (lane >> 4) * 8;  // C-layout row base: 0 or 8

    const float* hib = hi + (size_t)b * Tdim * Hdim;

    const v8f vzero = {0.f, 0.f, 0.f, 0.f, 0.f, 0.f, 0.f, 0.f};

    // ---------------- GEMM1: ah = hi[R0:R0+128, :] @ W_ah -------------------
    v8f acc[RFRAG][NTW];
    #pragma unroll
    for (int r = 0; r < RFRAG; ++r)
        #pragma unroll
        for (int i = 0; i < NTW; ++i) acc[r][i] = vzero;

    for (int ks = 0; ks < 16; ++ks) {
        v16h Bf[NTW];
        #pragma unroll
        for (int i = 0; i < NTW; ++i) {
            const int nt = wave + i * 16;
            Bf[i] = *(const v16h*)(W1pk + (((size_t)ks * 32 + nt) * 32 + lane) * 16);
        }
        #pragma unroll
        for (int r = 0; r < RFRAG; ++r) {
            const float* hr = hib + (size_t)(R0 + r * 16 + arow) * Hdim + ks * 32 + asel;
            v16h a;
            #pragma unroll
            for (int j = 0; j < 8; ++j) a[j]     = (_Float16)hr[j];
            #pragma unroll
            for (int j = 0; j < 8; ++j) a[8 + j] = (_Float16)hr[16 + j];
            #pragma unroll
            for (int i = 0; i < NTW; ++i)
                acc[r][i] = __builtin_amdgcn_wmma_f32_16x16x32_f16(
                                false, a, false, Bf[i], (short)0, acc[r][i], false, false);
        }
    }

    // ---------------- epilogue: x = tanh(ah + s) -> LDS (swizzled) ----------
    #pragma unroll
    for (int i = 0; i < NTW; ++i) {
        const int nt  = wave + i * 16;
        const int col = nt * 16 + (lane & 15);
        const float sb = s_in[b * Hdim + col];
        #pragma unroll
        for (int r = 0; r < RFRAG; ++r) {
            #pragma unroll
            for (int v = 0; v < 8; ++v) {
                const float xv = fast_tanh(acc[r][i][v] + sb);
                x_lds[xswz(r * 16 + rbase + v, col)] = (_Float16)xv;
            }
        }
    }
    __syncthreads();

    // ---------------- GEMM2: etk2 = x @ W_a ---------------------------------
    v8f acc2[RFRAG][NTW];
    #pragma unroll
    for (int r = 0; r < RFRAG; ++r)
        #pragma unroll
        for (int i = 0; i < NTW; ++i) acc2[r][i] = vzero;

    for (int ks = 0; ks < 16; ++ks) {
        v16h Bf[NTW];
        #pragma unroll
        for (int i = 0; i < NTW; ++i) {
            const int nt = wave + i * 16;
            Bf[i] = *(const v16h*)(W2pk + (((size_t)ks * 32 + nt) * 32 + lane) * 16);
        }
        const int o0 = ks * 32 + asel;
        #pragma unroll
        for (int r = 0; r < RFRAG; ++r) {
            const int rr = r * 16 + arow;
            const v8h c0 = *(const v8h*)(x_lds + xswz(rr, o0));
            const v8h c1 = *(const v8h*)(x_lds + xswz(rr, o0 + 16));
            v16h a;
            #pragma unroll
            for (int j = 0; j < 8; ++j) { a[j] = c0[j]; a[8 + j] = c1[j]; }
            #pragma unroll
            for (int i = 0; i < NTW; ++i)
                acc2[r][i] = __builtin_amdgcn_wmma_f32_16x16x32_f16(
                                 false, a, false, Bf[i], (short)0, acc2[r][i], false, false);
        }
    }

    // ---------------- partial softmax over this 128-row chunk ---------------
    // column k lives in lane pair (l, l^16); each lane holds 64 of 128 rows.
    #pragma unroll
    for (int i = 0; i < NTW; ++i) {
        const int nt = wave + i * 16;
        const int k  = nt * 16 + (lane & 15);
        float cm = acc2[0][i][0];
        #pragma unroll
        for (int r = 0; r < RFRAG; ++r)
            #pragma unroll
            for (int v = 0; v < 8; ++v) cm = fmaxf(cm, acc2[r][i][v]);
        cm = fmaxf(cm, __shfl_xor(cm, 16));               // full 128-row max
        float sp = 0.f, sn = 0.f;
        #pragma unroll
        for (int r = 0; r < RFRAG; ++r) {
            #pragma unroll
            for (int v = 0; v < 8; ++v) {
                const float p  = __expf(acc2[r][i][v] - cm);
                const float hv = hib[(size_t)(R0 + r * 16 + rbase + v) * Hdim + k];
                sp += p;
                sn  = fmaf(p, hv, sn);
            }
        }
        sp += __shfl_xor(sp, 16);
        sn += __shfl_xor(sn, 16);
        if (lane < 16) {
            const size_t idx = (size_t)wg * Hdim + k;
            cs_m[idx] = cm;
            cs_d[idx] = sp;
            cs_n[idx] = sn;
        }
    }
}

// ---------------------------------------------------------------------------
// Merge the chunk states per (b,k):  e = N/D with global rescaling.
// ---------------------------------------------------------------------------
__global__ void attn_merge(const float* __restrict__ cs_m, const float* __restrict__ cs_d,
                           const float* __restrict__ cs_n, float* __restrict__ out) {
    const int id = blockIdx.x * 256 + threadIdx.x;     // 0..131071 = b*512 + k
    const int b  = id >> 9;
    const int k  = id & 511;
    float m = -__builtin_inff();
    #pragma unroll
    for (int c = 0; c < NCHUNK; ++c)
        m = fmaxf(m, cs_m[(size_t)(b * NCHUNK + c) * Hdim + k]);
    float d = 0.f, n = 0.f;
    #pragma unroll
    for (int c = 0; c < NCHUNK; ++c) {
        const size_t idx = (size_t)(b * NCHUNK + c) * Hdim + k;
        const float sc = __expf(cs_m[idx] - m);
        d = fmaf(cs_d[idx], sc, d);
        n = fmaf(cs_n[idx], sc, n);
    }
    out[id] = n / d;
}

// ---------------------------------------------------------------------------
extern "C" void kernel_launch(void* const* d_in, const int* in_sizes, int n_in,
                              void* d_out, int out_size, void* d_ws, size_t ws_size,
                              hipStream_t stream) {
    const float* ht   = (const float*)d_in[0];   // [B,H]
    const float* ct   = (const float*)d_in[1];   // [B,H]
    const float* hi   = (const float*)d_in[2];   // [B,T,H]
    const float* W_as = (const float*)d_in[3];   // [2H,H]
    const float* W_ah = (const float*)d_in[4];   // [H,H]
    const float* ba   = (const float*)d_in[5];   // [1,H]
    // d_in[6] = v : unused by the reference computation
    const float* W_a  = (const float*)d_in[7];   // [H,H]
    float* out = (float*)d_out;                  // [B,H]

    char* ws = (char*)d_ws;
    float*    s_ws = (float*)ws;                          // 512 KB
    _Float16* W1pk = (_Float16*)(ws + (512 << 10));       // 512 KB
    _Float16* W2pk = (_Float16*)(ws + (1024 << 10));      // 512 KB
    float*    cs_m = (float*)(ws + (1536 << 10));         // 2 MB (1024*512*4)
    float*    cs_d = (float*)(ws + (1536 << 10) + (2 << 20));
    float*    cs_n = (float*)(ws + (1536 << 10) + (4 << 20));

    attn_phase1<<<Bdim, 512, 0, stream>>>(ht, ct, W_as, ba, s_ws);
    attn_pack_w<<<1024, 256, 0, stream>>>(W_ah, W1pk);
    attn_pack_w<<<1024, 256, 0, stream>>>(W_a,  W2pk);
    attn_chunk <<<Bdim * NCHUNK, 512, 0, stream>>>(hi, W1pk, W2pk, s_ws,
                                                   cs_m, cs_d, cs_n);
    attn_merge <<<Bdim * Hdim / 256, 256, 0, stream>>>(cs_m, cs_d, cs_n, out);
}
